// RWKV_65274912964788
// MI455X (gfx1250) — matrix operand order repaired and checked by reading, not compiled, not run on hardware
//
#include <hip/hip_runtime.h>

// ---------------------------------------------------------------------------
// RWKV forward for MI455X (gfx1250). ~175 GFLOP vs ~290 MB fp32 weight
// traffic -> HBM-bound (~12.5us floor at 23.3 TB/s). Stream fp32 weights once,
// convert to bf16 in-register, multiply with v_wmma_f32_16x16x32_bf16.
// GEMM is software-pipelined (double-buffered LDS) so global loads overlap
// WMMA compute instead of s_wait_loadcnt 0x0 after every load.
// ---------------------------------------------------------------------------

constexpr int T  = 1024;
constexpr int V  = 50304;
constexpr int C  = 768;
constexpr int H  = 12;
constexpr int L  = 6;
constexpr int DH = 64;            // d_head (K in the reference)
constexpr int F  = 2688;

typedef __attribute__((ext_vector_type(16))) __bf16 v16bf;
typedef __attribute__((ext_vector_type(8)))  float  v8f;

// ---------------------------------------------------------------------------
// WMMA fragment-order LDS addressing (CDNA5 ISA 7.12.2, wave32).
// A 16x32 bf16: lane = (k%16)/8*16 + m ; VGPR i = (k/16)*4 + (k%8)/2 ; j = k&1
//   -> for fixed m, k..k+3 (k%4==0) are CONTIGUOUS  => b64 LDS stores
// B 32x16 bf16: lane = (k%16)/8*16 + n ; elem e = (k/16)*8 + k%8
//   -> for fixed n, k..k+3 (k%4==0) are CONTIGUOUS  => b64 LDS stores
// ---------------------------------------------------------------------------
__device__ __forceinline__ int fragAddrA(int m, int k) {
  int group = k >> 4;
  int rem   = k & 15;
  int half  = rem >> 3;
  int rr    = rem & 7;
  int i     = group * 4 + (rr >> 1);
  int e     = 2 * i + (rr & 1);
  return (half * 16 + m) * 16 + e;
}
__device__ __forceinline__ int fragAddrB(int k, int n) {
  int khalf = (k >> 3) & 1;
  return (khalf * 16 + n) * 16 + (k >> 4) * 8 + (k & 7);
}

// ---------------------------------------------------------------------------
// GEMM: D[M,N] = op(A[M,Kd] @ B[Kd,N]) (+ Cadd). fp32 in, bf16 WMMA, f32 acc.
// Block 256 thr = 8 waves; tile 128x128x32; wave tile 32x64 (2x4 accums).
// Double-buffered LDS pipeline: load k+1 to regs | WMMA on k | cvt+store k+1.
// op: 0 = none, 2 = relu()^2
// ---------------------------------------------------------------------------
#define BM 128
#define BN 128
#define BK 32

__global__ void __launch_bounds__(256)
gemm_bf16_wmma(const float* __restrict__ A, const float* __restrict__ B,
               const float* __restrict__ Cadd, float* __restrict__ D,
               int M, int N, int Kd, int op) {
  __shared__ __align__(32) __bf16 As[2][8 * 512];
  __shared__ __align__(32) __bf16 Bs[2][8 * 512];

  const int tid  = threadIdx.x;
  const int lane = tid & 31;
  const int wave = tid >> 5;
  const int wm   = wave >> 1;        // 0..3 -> m offset wm*32
  const int wn   = wave & 1;         // 0..1 -> n offset wn*64
  const int bM   = blockIdx.y * BM;
  const int bN   = blockIdx.x * BN;

  // ---- per-thread K-invariant addressing (hoisted out of the K loop) ----
  // A tile: chunk c = tid+i*256 -> row = c>>3 (0..127), k4 = (c&7)*4
  const float4* aPtr[4]; int aOff[4];
#pragma unroll
  for (int i = 0; i < 4; ++i) {
    int c = tid + i * 256, row = c >> 3, k4 = (c & 7) * 4;
    aPtr[i] = reinterpret_cast<const float4*>(A + (size_t)(bM + row) * Kd + k4);
    aOff[i] = (row >> 4) * 512 + fragAddrA(row & 15, k4);
  }
  // B tile: chunk c -> n = c&127, kq = (c>>7)*4 ; 4 scalar loads along k,
  // coalesced across threads along n; contiguous bf16 quad in LDS.
  const float* bPtr[4]; int bOff[4];
#pragma unroll
  for (int i = 0; i < 4; ++i) {
    int c = tid + i * 256, n = c & 127, kq = (c >> 7) * 4;
    bPtr[i] = B + (size_t)kq * N + bN + n;
    bOff[i] = (n >> 4) * 512 + fragAddrB(kq, n & 15);
  }
  const size_t bStepK = (size_t)BK * N;       // floats per K-tile of B

  v8f acc[2][4];
#pragma unroll
  for (int i = 0; i < 2; ++i)
#pragma unroll
    for (int j = 0; j < 4; ++j)
#pragma unroll
      for (int q = 0; q < 8; ++q) acc[i][j][q] = 0.0f;

  float4 fa[4];
  float  fb[4][4];

  auto g_load = [&](int kt) {                 // issue 4x b128 + 16x b32 loads
#pragma unroll
    for (int i = 0; i < 4; ++i) fa[i] = aPtr[i][(size_t)kt * 8];
#pragma unroll
    for (int i = 0; i < 4; ++i)
#pragma unroll
      for (int j = 0; j < 4; ++j)
        fb[i][j] = bPtr[i][(size_t)kt * bStepK + (size_t)j * N];
  };
  auto lds_store = [&](int buf) {             // cvt to bf16, b64-packable
#pragma unroll
    for (int i = 0; i < 4; ++i) {
      __bf16* p = &As[buf][aOff[i]];
      p[0] = (__bf16)fa[i].x; p[1] = (__bf16)fa[i].y;
      p[2] = (__bf16)fa[i].z; p[3] = (__bf16)fa[i].w;
    }
#pragma unroll
    for (int i = 0; i < 4; ++i) {
      __bf16* p = &Bs[buf][bOff[i]];
      p[0] = (__bf16)fb[i][0]; p[1] = (__bf16)fb[i][1];
      p[2] = (__bf16)fb[i][2]; p[3] = (__bf16)fb[i][3];
    }
  };
  auto compute = [&](int buf) {
    v16bf bfr[4];
#pragma unroll
    for (int nf = 0; nf < 4; ++nf)
      bfr[nf] = *reinterpret_cast<const v16bf*>(&Bs[buf][(wn * 4 + nf) * 512 + lane * 16]);
#pragma unroll
    for (int mf = 0; mf < 2; ++mf) {
      v16bf af = *reinterpret_cast<const v16bf*>(&As[buf][(wm * 2 + mf) * 512 + lane * 16]);
#pragma unroll
      for (int nf = 0; nf < 4; ++nf)
        acc[mf][nf] = __builtin_amdgcn_wmma_f32_16x16x32_bf16(
            false, af, false, bfr[nf], (short)0, acc[mf][nf], false, false);
    }
  };

  const int nk = Kd / BK;
  g_load(0);
  lds_store(0);
  __syncthreads();
  for (int kt = 0; kt < nk; ++kt) {
    const int cur = kt & 1;
    if (kt + 1 < nk) {
      g_load(kt + 1);                         // in flight during WMMA below
      if (kt + 2 < nk) {                      // global_prefetch_b8 of k+2
        __builtin_prefetch(aPtr[0] + (size_t)(kt + 2) * 8, 0, 1);
        __builtin_prefetch(aPtr[2] + (size_t)(kt + 2) * 8, 0, 1);
        __builtin_prefetch(bPtr[0] + (size_t)(kt + 2) * bStepK, 0, 1);
      }
    }
    compute(cur);
    if (kt + 1 < nk) lds_store(cur ^ 1);      // waits loadcnt here, not at load
    __syncthreads();
  }

  // ---- epilogue: lane holds col = lane%16, rows q + 8*(lane/16) ----
  const int cbase = bN + wn * 64;
  const int rbase = bM + wm * 32;
#pragma unroll
  for (int mf = 0; mf < 2; ++mf) {
#pragma unroll
    for (int nf = 0; nf < 4; ++nf) {
      int col    = cbase + nf * 16 + (lane & 15);
      int rowoff = rbase + mf * 16 + ((lane >> 4) * 8);
#pragma unroll
      for (int q = 0; q < 8; ++q) {
        size_t idx = (size_t)(rowoff + q) * N + col;
        float val = acc[mf][nf][q];
        if (op == 2) { val = fmaxf(val, 0.0f); val = val * val; }
        if (Cadd) val += Cadd[idx];
        D[idx] = val;
      }
    }
  }
}

// ---------------------------------------------------------------------------
// LayerNorm helpers
// ---------------------------------------------------------------------------
__device__ __forceinline__ void block_sum2_256(float& s, float& ss) {
  __shared__ float bs[256], bss[256];
  int tid = threadIdx.x;
  bs[tid] = s; bss[tid] = ss;
  __syncthreads();
  for (int off = 128; off > 0; off >>= 1) {
    if (tid < off) { bs[tid] += bs[tid + off]; bss[tid] += bss[tid + off]; }
    __syncthreads();
  }
  s = bs[0]; ss = bss[0];
  __syncthreads();
}

__device__ __forceinline__ void ln_row(const float* __restrict__ row,
                                       const float* __restrict__ wb,
                                       float* __restrict__ out, float eps) {
  float s = 0.f, ss = 0.f;
  for (int c = threadIdx.x; c < C; c += 256) { float v = row[c]; s += v; ss += v * v; }
  block_sum2_256(s, ss);
  float mu  = s * (1.0f / C);
  float var = ss * (1.0f / C) - mu * mu;
  float inv = rsqrtf(var + eps);
  for (int c = threadIdx.x; c < C; c += 256)
    out[c] = (row[c] - mu) * inv * wb[c] + wb[C + c];
}

__global__ void __launch_bounds__(256)
embed_ln_kernel(const int* __restrict__ tokens, const float* __restrict__ embed,
                const float* __restrict__ wb, float* __restrict__ y) {
  int t = blockIdx.x;
  ln_row(embed + (size_t)tokens[t] * C, wb, y + (size_t)t * C, 1e-5f);
}

__global__ void __launch_bounds__(256)
ln_kernel(const float* __restrict__ x, const float* __restrict__ wb,
          float* __restrict__ y) {
  int t = blockIdx.x;
  ln_row(x + (size_t)t * C, wb, y + (size_t)t * C, 1e-5f);
}

// token-shift + lerp: out[j][t][c] = xp + (xn - xp) * ts[j][c], xp = xn[t-1] (0 at t=0)
__global__ void __launch_bounds__(256)
lerp_kernel(const float* __restrict__ xn, const float* __restrict__ ts,
            float* __restrict__ out, int nmix) {
  int t = blockIdx.x;
  for (int c = threadIdx.x; c < C; c += 256) {
    float cur  = xn[(size_t)t * C + c];
    float prev = (t > 0) ? xn[(size_t)(t - 1) * C + c] : 0.0f;
    for (int j = 0; j < nmix; ++j)
      out[(size_t)j * T * C + (size_t)t * C + c] = prev + (cur - prev) * ts[j * C + c];
  }
}

// ---------------------------------------------------------------------------
// WKV scan: one block per head (12 blocks, 64 threads). Thread owns column v
// of the 64x64 state in registers. Per-step r/k broadcast vectors are staged
// into LDS with gfx1250 async global->LDS copies (ASYNCcnt path).
// ---------------------------------------------------------------------------
__global__ void __launch_bounds__(64)
wkv_kernel(const float* __restrict__ r, const float* __restrict__ k,
           const float* __restrict__ v, const float* __restrict__ bonus,
           const float* __restrict__ decay, float* __restrict__ o) {
  const int h   = blockIdx.x;
  const int tid = threadIdx.x;                  // v-index
  __shared__ float rS[DH], kS[DH];
  float wR[DH], bR[DH], kv[DH];
#pragma unroll
  for (int q = 0; q < DH; ++q) {
    wR[q] = __expf(-__expf(decay[h * DH + q]));
    bR[q] = bonus[h * DH + q];
    kv[q] = 0.0f;
  }
  // generic LDS pointer low 32 bits == LDS byte offset (ISA 10.2 aperture rule)
  unsigned ldsr = (unsigned)(unsigned long long)&rS[tid];
  unsigned ldsk = (unsigned)(unsigned long long)&kS[tid];
  for (int t = 0; t < T; ++t) {
    size_t base = (size_t)t * C + h * DH;
    const float* gr = r + base + tid;
    const float* gk = k + base + tid;
    asm volatile("global_load_async_to_lds_b32 %0, %1, off"
                 :: "v"(ldsr), "v"(gr) : "memory");
    asm volatile("global_load_async_to_lds_b32 %0, %1, off"
                 :: "v"(ldsk), "v"(gk) : "memory");
    float vt = v[base + tid];
    asm volatile("s_wait_asynccnt 0x0" ::: "memory");
    __syncthreads();
    float out = 0.0f;
#pragma unroll
    for (int q = 0; q < DH; ++q) {
      float a = kS[q] * vt;                      // outer-product column
      out  += rS[q] * __builtin_fmaf(a, bR[q], kv[q]);
      kv[q] = __builtin_fmaf(kv[q], wR[q], a);
    }
    o[base + tid] = out;
    __syncthreads();
  }
}

// groupnorm (per 64-chan head group) then * silu(g); one block per (t,h)
__global__ void __launch_bounds__(64)
gn_silu_kernel(const float* __restrict__ xin, const float* __restrict__ g,
               const float* __restrict__ wb, float* __restrict__ y) {
  int blk = blockIdx.x;
  int t = blk / H, h = blk % H;
  int tid = threadIdx.x;
  int ci  = h * DH + tid;
  size_t idx = (size_t)t * C + ci;
  float val = xin[idx];
  __shared__ float bs[64], bss[64];
  bs[tid] = val; bss[tid] = val * val;
  __syncthreads();
  for (int off = 32; off > 0; off >>= 1) {
    if (tid < off) { bs[tid] += bs[tid + off]; bss[tid] += bss[tid + off]; }
    __syncthreads();
  }
  float mu  = bs[0] * (1.0f / DH);
  float var = bss[0] * (1.0f / DH) - mu * mu;
  float o   = (val - mu) * rsqrtf(var + 6.4e-4f) * wb[ci] + wb[C + ci];
  float gg  = g[idx];
  y[idx] = o * (gg / (1.0f + __expf(-gg)));      // silu gate
}

// x += hw * sigmoid(gate)
__global__ void __launch_bounds__(256)
gate_add_kernel(float* __restrict__ x, const float* __restrict__ hw,
                const float* __restrict__ gate, int n) {
  int i = blockIdx.x * 256 + threadIdx.x;
  if (i < n) {
    float gg = gate[i];
    x[i] += hw[i] * (1.0f / (1.0f + __expf(-gg)));
  }
}

// ---------------------------------------------------------------------------
extern "C" void kernel_launch(void* const* d_in, const int* in_sizes, int n_in,
                              void* d_out, int out_size, void* d_ws, size_t ws_size,
                              hipStream_t stream) {
  (void)in_sizes; (void)n_in; (void)out_size; (void)ws_size;
  const int*   tokens   = (const int*)d_in[0];
  const float* embed    = (const float*)d_in[1];
  const float* embed_ln = (const float*)d_in[2];
  const float* tm_ln    = (const float*)d_in[3];
  const float* tm_ts    = (const float*)d_in[4];
  const float* tm_Wr    = (const float*)d_in[5];
  const float* tm_Wk    = (const float*)d_in[6];
  const float* tm_Wv    = (const float*)d_in[7];
  const float* tm_Wg    = (const float*)d_in[8];
  const float* tm_Wo    = (const float*)d_in[9];
  const float* tm_bonus = (const float*)d_in[10];
  const float* tm_decay = (const float*)d_in[11];
  const float* tm_gn    = (const float*)d_in[12];
  const float* cm_ln    = (const float*)d_in[13];
  const float* cm_ts    = (const float*)d_in[14];
  const float* cm_Win   = (const float*)d_in[15];
  const float* cm_Wout  = (const float*)d_in[16];
  const float* cm_Wg    = (const float*)d_in[17];
  const float* head_ln  = (const float*)d_in[18];
  const float* unembed  = (const float*)d_in[19];
  float* out = (float*)d_out;

  const size_t TC = (size_t)T * C, TF = (size_t)T * F;
  float* ws   = (float*)d_ws;
  float* x    = ws;            // [T,C]
  float* xn   = x    + TC;     // [T,C]
  float* mix  = xn   + TC;     // [4,T,C]
  float* rb   = mix  + 4 * TC; // [T,C]
  float* kb   = rb   + TC;
  float* vb   = kb   + TC;
  float* gb   = vb   + TC;
  float* wkvo = gb   + TC;
  float* gno  = wkvo + TC;
  float* hf   = gno  + TC;     // [T,F]
  float* gate = hf   + TF;     // [T,C]
  float* hw   = gate + TC;     // [T,C]

  const dim3 gC(C / BN, T / BM);
  const dim3 gF(F / BN, T / BM);
  const dim3 gV(V / BN, T / BM);

  embed_ln_kernel<<<T, 256, 0, stream>>>(tokens, embed, embed_ln, x);

  for (int l = 0; l < L; ++l) {
    // ---- time mixer ----
    ln_kernel<<<T, 256, 0, stream>>>(x, tm_ln + (size_t)l * 2 * C, xn);
    lerp_kernel<<<T, 256, 0, stream>>>(xn, tm_ts + (size_t)l * 4 * C, mix, 4);
    gemm_bf16_wmma<<<gC, 256, 0, stream>>>(mix + 0 * TC, tm_Wr + (size_t)l * C * C, nullptr, rb, T, C, C, 0);
    gemm_bf16_wmma<<<gC, 256, 0, stream>>>(mix + 1 * TC, tm_Wk + (size_t)l * C * C, nullptr, kb, T, C, C, 0);
    gemm_bf16_wmma<<<gC, 256, 0, stream>>>(mix + 2 * TC, tm_Wv + (size_t)l * C * C, nullptr, vb, T, C, C, 0);
    gemm_bf16_wmma<<<gC, 256, 0, stream>>>(mix + 3 * TC, tm_Wg + (size_t)l * C * C, nullptr, gb, T, C, C, 0);
    wkv_kernel<<<H, 64, 0, stream>>>(rb, kb, vb, tm_bonus + (size_t)l * H * DH,
                                     tm_decay + (size_t)l * H * DH, wkvo);
    gn_silu_kernel<<<T * H, 64, 0, stream>>>(wkvo, gb, tm_gn + (size_t)l * 2 * C, gno);
    gemm_bf16_wmma<<<gC, 256, 0, stream>>>(gno, tm_Wo + (size_t)l * C * C, x, x, T, C, C, 0); // +residual
    // ---- channel mixer ----
    ln_kernel<<<T, 256, 0, stream>>>(x, cm_ln + (size_t)l * 2 * C, xn);
    lerp_kernel<<<T, 256, 0, stream>>>(xn, cm_ts + (size_t)l * 2 * C, mix, 2);
    gemm_bf16_wmma<<<gF, 256, 0, stream>>>(mix, cm_Win + (size_t)l * C * F, nullptr, hf, T, F, C, 2); // relu^2
    gemm_bf16_wmma<<<gC, 256, 0, stream>>>(mix + TC, cm_Wg + (size_t)l * C * C, nullptr, gate, T, C, C, 0);
    gemm_bf16_wmma<<<gC, 256, 0, stream>>>(hf, cm_Wout + (size_t)l * F * C, nullptr, hw, T, C, F, 0);
    gate_add_kernel<<<(int)(TC / 256), 256, 0, stream>>>(x, hw, gate, (int)TC);
  }

  ln_kernel<<<T, 256, 0, stream>>>(x, head_ln, xn);
  gemm_bf16_wmma<<<gV, 256, 0, stream>>>(xn, unembed, nullptr, out, T, V, C, 0);
}
